// MPDOOpen_86732569576139
// MI455X (gfx1250) — matrix operand order repaired and checked by reading, not compiled, not run on hardware
//
#include <hip/hip_runtime.h>
#include <math.h>

typedef __attribute__((ext_vector_type(16))) _Float16 v16h;
typedef __attribute__((ext_vector_type(8)))  _Float16 v8h;
typedef __attribute__((ext_vector_type(8)))  float    v8f;

#define LSITES 64     // L
#define NMID   62     // L-2 middle sites
#define DM     32     // bond dim D
#define PITCH  40     // row pitch in halfs: 80B -> 16B-aligned rows, conflict-free banks
#define NW     4      // waves per block
#define NBATCH 2048

// Assemble a WMMA 16-half operand from two 16-byte LDS chunks.
static __device__ __forceinline__ v16h ldop(const _Float16* p0, const _Float16* p1) {
    v8h lo = *(const v8h*)p0;
    v8h hi = *(const v8h*)p1;
    v16h r;
#pragma unroll
    for (int i = 0; i < 8; ++i) { r[i] = lo[i]; r[i + 8] = hi[i]; }
    return r;
}

__global__ __launch_bounds__(32 * NW) void mpdo_wmma_kernel(
    const int*   __restrict__ qn,    // [B][2L]
    const float* __restrict__ lt,    // [d][D][X]
    const float* __restrict__ rt,    // [d][D][X]
    const float* __restrict__ mid,   // [L-2][d][D][D][X]
    float*       __restrict__ out)   // [B] complex64 -> interleaved (re,im)
{
    // Shared per-site tensor stage: sA[s][k][c*PITCH + r] = mid[t][s][r][c][k]
    //  = row-major A_s[:,:,k]^T   (phase-1 A-operand)
    //  = col-major A_s[:,:,k]     (phase-2 B-operand)
    __shared__ __align__(16) _Float16 sA[2][2][DM * PITCH];
    // Per-wave edge, column-major: sE[n*PITCH + r] = E[r][n]  (phase-1 B-operand)
    __shared__ __align__(16) _Float16 sE[NW][DM * PITCH];
    // Per-wave T_k, row-major: sT[k][m*PITCH + kk] = T_k[m][kk] (phase-2 A-operand)
    __shared__ __align__(16) _Float16 sT[NW][2][DM * PITCH];

    const int tid  = threadIdx.x;
    const int w    = tid >> 5;
    const int lane = tid & 31;
    const int half = lane >> 4;
    const int l16  = lane & 15;
    const int b    = blockIdx.x * NW + w;

    const int* q = qn + b * (2 * LSITES);
    const int klo = 8 * half;    // A-operand K start (per half-wave)
    const int kb  = 16 * half;   // unused: B operands also start at klo

    // ---- initial edge: E0[i][l] = sum_x lt[sr0][i][x] * lt[sc0][l][x]
    {
        const int sr0 = q[0], sc0 = q[LSITES];
        const float* Lr = lt + sr0 * DM * 2;
        const float* Lc = lt + sc0 * DM * 2;
        const float r0 = Lr[lane * 2 + 0], r1 = Lr[lane * 2 + 1];
        for (int l = 0; l < DM; ++l) {
            float v = r0 * Lc[l * 2 + 0] + r1 * Lc[l * 2 + 1];
            sE[w][l * PITCH + lane] = (_Float16)v;   // col-major store
        }
    }

    for (int t = 0; t < NMID; ++t) {
        __syncthreads();   // all waves done reading sA from previous site
        // ---- cooperative stage of mid[t] (4096 floats) into sA as f16
        {
            const float* mt = mid + t * 4096;
#pragma unroll
            for (int u = 0; u < 16; ++u) {
                int e   = tid + u * 128;        // 0..2047 float2 elements
                int s   = e >> 10;
                int rem = e & 1023;
                int i   = rem >> 5;
                int j   = rem & 31;
                const float* p = mt + s * 2048 + i * 64 + j * 2;  // contiguous kraus pair
                float k0 = p[0], k1 = p[1];
                sA[s][0][j * PITCH + i] = (_Float16)k0;
                sA[s][1][j * PITCH + i] = (_Float16)k1;
            }
        }
        __syncthreads();   // staging visible to all waves

        const int sr = q[1 + t];
        const int sc = q[LSITES + 1 + t];

        // ---- phase 1: T_k = A_sr[:,:,k]^T @ E   (2k x 2x2 tiles, K=32 each)
        v16h bop[2];
#pragma unroll
        for (int bb = 0; bb < 2; ++bb) {           // E columns (shared across k)
            const _Float16* base = &sE[w][(bb * 16 + l16) * PITCH];
            bop[bb] = ldop(base + klo, base + 16 + klo);
        }
#pragma unroll
        for (int k = 0; k < 2; ++k) {
            v16h aop[2];
#pragma unroll
            for (int aa = 0; aa < 2; ++aa) {       // rows of A^T
                const _Float16* base = &sA[sr][k][(aa * 16 + l16) * PITCH];
                aop[aa] = ldop(base + klo, base + 16 + klo);
            }
#pragma unroll
            for (int aa = 0; aa < 2; ++aa)
#pragma unroll
            for (int bb = 0; bb < 2; ++bb) {
                v8f acc = {};
                acc = __builtin_amdgcn_wmma_f32_16x16x32_f16(
                        false, aop[aa], false, bop[bb], (short)0, acc, false, false);
                const int n = bb * 16 + l16;       // scatter C-fragments to row-major T
#pragma unroll
                for (int v = 0; v < 8; ++v) {
                    int m = aa * 16 + v + 8 * half;
                    sT[w][k][m * PITCH + n] = (_Float16)acc[v];
                }
            }
        }

        // ---- phase 2: E' = sum_k T_k @ A_sc[:,:,k]  (2x2 tiles, 2 chained WMMAs)
#pragma unroll
        for (int aa = 0; aa < 2; ++aa)
#pragma unroll
        for (int bb = 0; bb < 2; ++bb) {
            v8f acc = {};
#pragma unroll
            for (int k = 0; k < 2; ++k) {
                const _Float16* ab = &sT[w][k][(aa * 16 + l16) * PITCH];
                v16h a = ldop(ab + klo, ab + 16 + klo);
                const _Float16* bbse = &sA[sc][k][(bb * 16 + l16) * PITCH];
                v16h bo = ldop(bbse + klo, bbse + 16 + klo);
                acc = __builtin_amdgcn_wmma_f32_16x16x32_f16(
                        false, a, false, bo, (short)0, acc, false, false);
            }
            // store E' column-major, rescaled by 0.5 to keep f16 in range
            const int n = bb * 16 + l16;
            v8h pk;
#pragma unroll
            for (int v = 0; v < 8; ++v) pk[v] = (_Float16)(0.5f * acc[v]);
            *(v8h*)(&sE[w][n * PITCH + aa * 16 + 8 * half]) = pk;
        }
    }

    // ---- final: rho = sum_{i,j} E[i][j] * (rt[srL] @ rt[scL]^T)[i][j]
    {
        const int srL = q[LSITES - 1], scL = q[2 * LSITES - 1];
        const float* Rr = rt + srL * DM * 2;
        const float* Rc = rt + scL * DM * 2;
        const float rr0 = Rr[lane * 2 + 0], rr1 = Rr[lane * 2 + 1];
        float acc = 0.f;
        for (int j = 0; j < DM; ++j) {
            float e = (float)sE[w][j * PITCH + lane];          // E[lane][j]
            float r = rr0 * Rc[j * 2 + 0] + rr1 * Rc[j * 2 + 1];
            acc += e * r;
        }
#pragma unroll
        for (int off = 16; off > 0; off >>= 1)
            acc += __shfl_xor(acc, off, 32);
        if (lane == 0) {
            // undo the 62 per-step 0.5 rescales: log(rho_true) = log(rho) + 62*ln2
            float re = logf(fabsf(acc)) + (float)NMID * 0.69314718055994530942f;
            float im = (acc < 0.f) ? 3.14159265358979323846f : 0.f;
            out[2 * b + 0] = re;
            out[2 * b + 1] = im;
        }
    }
}

extern "C" void kernel_launch(void* const* d_in, const int* in_sizes, int n_in,
                              void* d_out, int out_size, void* d_ws, size_t ws_size,
                              hipStream_t stream) {
    (void)in_sizes; (void)n_in; (void)out_size; (void)d_ws; (void)ws_size;
    const int*   qn  = (const int*)d_in[0];
    const float* lt  = (const float*)d_in[1];
    const float* rt  = (const float*)d_in[2];
    const float* mid = (const float*)d_in[3];
    float* out = (float*)d_out;

    dim3 grid(NBATCH / NW);
    dim3 block(32 * NW);
    hipLaunchKernelGGL(mpdo_wmma_kernel, grid, block, 0, stream,
                       qn, lt, rt, mid, out);
}
